// TransformerLayer_30846455120626
// MI455X (gfx1250) — compile-verified, matrix operand order checked by reading, not run on hardware
//
#include <hip/hip_runtime.h>
#include <hip/hip_bf16.h>
#include <math.h>

typedef __attribute__((ext_vector_type(16))) __bf16 v16bf;
typedef __attribute__((ext_vector_type(8)))  __bf16 v8bf;
typedef __attribute__((ext_vector_type(8)))  float  v8f;
typedef __attribute__((ext_vector_type(4)))  int    v4i;

#define DEV static __device__ __forceinline__
#define GAS __attribute__((address_space(1)))
#define LAS __attribute__((address_space(3)))

#if __has_builtin(__builtin_amdgcn_global_load_async_to_lds_b128) && \
    __has_builtin(__builtin_amdgcn_s_wait_asynccnt)
#define HAS_ASYNC_LDS 1
#else
#define HAS_ASYNC_LDS 0
#endif

namespace {
constexpr int   TB   = 2;
constexpr int   TS   = 2048;
constexpr int   TD   = 2048;
constexpr int   THD  = 16;     // heads
constexpr int   TDH  = 128;    // head dim
constexpr int   TDFF = 8192;
constexpr int   TM   = TB * TS;          // 4096 token rows
constexpr float TEPS = 1e-5f;
}

// ---------- scalar helpers ----------
DEV __bf16 f2bf(float x) {
  unsigned u = __builtin_bit_cast(unsigned, x);
  unsigned r = u + 0x7FFFu + ((u >> 16) & 1u);      // RNE
  unsigned short s = (unsigned short)(r >> 16);
  return __builtin_bit_cast(__bf16, s);
}

DEV float gelu_f(float x) {
  const float c = 0.7978845608028654f;
  float x3 = x * x * x;
  return 0.5f * x * (1.f + tanhf(c * (x + 0.044715f * x3)));
}

// 16B global -> LDS copy: async-to-LDS (ASYNCcnt) when available, else
// load + ds_store fallback. Builtin signature (from hipcc diagnostic):
// (AS1 v4i* global_src, AS3 v4i* lds_dst, imm offset, imm cpol).
DEV void cp16_g2l(const __bf16* g, __bf16* l) {
#if HAS_ASYNC_LDS
  __builtin_amdgcn_global_load_async_to_lds_b128(
      (GAS v4i*)g, (LAS v4i*)l, 0, 0);
#else
  *(uint4*)l = *(const uint4*)g;
#endif
}
DEV void async_wait0() {
#if HAS_ASYNC_LDS
  __builtin_amdgcn_s_wait_asynccnt(0);
#endif
}

// Load one 16x(32-chunk) bf16 fragment (A-style row layout).
// Per the CDNA5 ISA layout, lane L holds row (L&15); its 16 bf16 elements are
// two contiguous 8-element runs at K offsets {8*(L>>4), 16+8*(L>>4)}.
DEV v16bf frag_ld(const __bf16* rowbase, int lhi) {
  v8bf lo = *(const v8bf*)(rowbase + 8 * lhi);
  v8bf hi = *(const v8bf*)(rowbase + 16 + 8 * lhi);
  return __builtin_shufflevector(lo, hi, 0, 1, 2, 3, 4, 5, 6, 7,
                                         8, 9, 10, 11, 12, 13, 14, 15);
}

DEV v8f wmma_bf16(v16bf a, v16bf b, v8f c) {
  return __builtin_amdgcn_wmma_f32_16x16x32_bf16(false, a, false, b,
                                                 (short)0, c, false, false);
}

// ---------- LayerNorm: f32 in -> bf16 out ----------
__global__ __launch_bounds__(256) void layernorm_bf16_k(
    const float* __restrict__ X, const float* __restrict__ w,
    const float* __restrict__ b, __bf16* __restrict__ out) {
  __shared__ float r1[256];
  __shared__ float r2[256];
  const int row = blockIdx.x;
  const int tid = threadIdx.x;
  const float* x = X + (size_t)row * TD;
  float lx[8], s = 0.f, ss = 0.f;
#pragma unroll
  for (int i = 0; i < 8; i++) {
    lx[i] = x[tid + i * 256];
    s += lx[i];
    ss += lx[i] * lx[i];
  }
  r1[tid] = s; r2[tid] = ss;
  __syncthreads();
  for (int off = 128; off > 0; off >>= 1) {
    if (tid < off) { r1[tid] += r1[tid + off]; r2[tid] += r2[tid + off]; }
    __syncthreads();
  }
  const float mean = r1[0] * (1.f / TD);
  const float var  = r2[0] * (1.f / TD) - mean * mean;
  const float rstd = rsqrtf(var + TEPS);
#pragma unroll
  for (int i = 0; i < 8; i++) {
    const int col = tid + i * 256;
    out[(size_t)row * TD + col] = f2bf((lx[i] - mean) * rstd * w[col] + b[col]);
  }
}

// ---------- Tiled WMMA GEMM:  C[M,N] = A[M,K](bf16) @ W[N,K]^T(f32) + bias ----------
// Double-buffered LDS, one barrier per 32-deep K step, prefetch 2 tiles ahead.
// OUTM: 0 = bf16 [m][n], 1 = f32 [m][n], 2 = bf16 head-transposed V_t[b][h][dh][s]
template <int OUTM, bool GELU, bool RES>
__global__ __launch_bounds__(256) void gemm_bf16_k(
    const __bf16* __restrict__ A, const float* __restrict__ W,
    const float* __restrict__ bias, const float* __restrict__ res,
    void* __restrict__ outp, int M, int N, int K, float scale) {
  constexpr int SA = 40;                    // 32 + 8 pad (bf16 elems)
  __shared__ __bf16 As[2][128 * SA];
  __shared__ __bf16 Bs[2][128 * SA];
  const int tid = threadIdx.x, lane = tid & 31, wave = tid >> 5;
  const int lrow = lane & 15, lhi = (lane >> 4) & 1;
  const int m0 = blockIdx.y * 128, n0 = blockIdx.x * 128;
  const int mb = (wave & 3) * 32, nb = (wave >> 2) * 64;
  const int srow = tid >> 1, scol = (tid & 1) * 16;

  v8f acc[2][4];
#pragma unroll
  for (int i = 0; i < 2; i++)
#pragma unroll
    for (int j = 0; j < 4; j++)
#pragma unroll
      for (int e = 0; e < 8; e++) acc[i][j][e] = 0.f;

  auto stage = [&](int buf, int k0) {
    // A tile: bf16 pass-through -> async global->LDS when available
    const __bf16* asrc = A + (size_t)(m0 + srow) * K + k0 + scol;
    cp16_g2l(asrc,     &As[buf][srow * SA + scol]);
    cp16_g2l(asrc + 8, &As[buf][srow * SA + scol + 8]);
    // W tile: f32 -> bf16 convert in flight (manual path)
    const float* wsrc = W + (size_t)(n0 + srow) * K + k0 + scol;
    float4 f0 = *(const float4*)(wsrc);
    float4 f1 = *(const float4*)(wsrc + 4);
    float4 f2 = *(const float4*)(wsrc + 8);
    float4 f3 = *(const float4*)(wsrc + 12);
    union { __bf16 h8[8]; uint4 q; } u0, u1;
    u0.h8[0] = f2bf(f0.x); u0.h8[1] = f2bf(f0.y); u0.h8[2] = f2bf(f0.z); u0.h8[3] = f2bf(f0.w);
    u0.h8[4] = f2bf(f1.x); u0.h8[5] = f2bf(f1.y); u0.h8[6] = f2bf(f1.z); u0.h8[7] = f2bf(f1.w);
    u1.h8[0] = f2bf(f2.x); u1.h8[1] = f2bf(f2.y); u1.h8[2] = f2bf(f2.z); u1.h8[3] = f2bf(f2.w);
    u1.h8[4] = f2bf(f3.x); u1.h8[5] = f2bf(f3.y); u1.h8[6] = f2bf(f3.z); u1.h8[7] = f2bf(f3.w);
    *(uint4*)(&Bs[buf][srow * SA + scol])     = u0.q;
    *(uint4*)(&Bs[buf][srow * SA + scol + 8]) = u1.q;
  };

  stage(0, 0);
  int buf = 0;
  for (int k0 = 0; k0 < K; k0 += 32, buf ^= 1) {
    async_wait0();            // writes into As[buf] from this wave are done
    __syncthreads();          // all waves' stage of [buf] complete
    if (k0 + 32 < K) {
      if (k0 + 64 < K) {      // pull the tile after next into cache
        __builtin_prefetch(A + (size_t)(m0 + srow) * K + k0 + 64 + scol, 0, 1);
        __builtin_prefetch(W + (size_t)(n0 + srow) * K + k0 + 64 + scol, 0, 1);
      }
      stage(buf ^ 1, k0 + 32);  // overlap with WMMAs below
    }
    v16bf af[2], bfr[4];
#pragma unroll
    for (int mi = 0; mi < 2; mi++)
      af[mi] = frag_ld(&As[buf][(mb + mi * 16 + lrow) * SA], lhi);
#pragma unroll
    for (int ni = 0; ni < 4; ni++)
      bfr[ni] = frag_ld(&Bs[buf][(nb + ni * 16 + lrow) * SA], lhi);
#pragma unroll
    for (int mi = 0; mi < 2; mi++)
#pragma unroll
      for (int ni = 0; ni < 4; ni++)
        acc[mi][ni] = wmma_bf16(af[mi], bfr[ni], acc[mi][ni]);
  }

  // epilogue: C/D layout -> lane holds (m = v + 8*lhi, n = lane&15) per frag
#pragma unroll
  for (int mi = 0; mi < 2; mi++) {
#pragma unroll
    for (int ni = 0; ni < 4; ni++) {
      const int n = n0 + nb + ni * 16 + lrow;
      const float bv = bias[n];
#pragma unroll
      for (int v = 0; v < 8; v++) {
        const int m = m0 + mb + mi * 16 + v + 8 * lhi;
        float t = (acc[mi][ni][v] + bv) * scale;
        if (GELU) t = gelu_f(t);
        if (RES)  t += res[(size_t)m * N + n];
        if (OUTM == 0) {
          ((__bf16*)outp)[(size_t)m * N + n] = f2bf(t);
        } else if (OUTM == 1) {
          ((float*)outp)[(size_t)m * N + n] = t;
        } else {  // V_t[b][h][dh][s]
          const int b_ = m / TS, s = m % TS;
          const int h = n / TDH, dh = n % TDH;
          ((__bf16*)outp)[((size_t)(b_ * THD + h) * TDH + dh) * TS + s] = f2bf(t);
        }
      }
    }
  }
}

// ---------- Flash attention (causal), one (b,h) x 128-query block per WG ----------
// Computes S^T = K Q^T so query == lane column; O^T = V^T P^T with V_t rows.
__global__ __launch_bounds__(256) void attn_k(
    const __bf16* __restrict__ Qb, const __bf16* __restrict__ Kb,
    const __bf16* __restrict__ Vt, __bf16* __restrict__ Ob) {
  constexpr int SK = 136, SV = 40, SP = 40;
  __shared__ __bf16 Ks[32 * SK];    // [key][d]
  __shared__ __bf16 Vs[128 * SV];   // [d][key]
  __shared__ __bf16 Ps[8 * 16 * SP];// per-wave P[q][key]
  const int tid = threadIdx.x, lane = tid & 31, wave = tid >> 5;
  const int lrow = lane & 15, lhi = (lane >> 4) & 1;
  const int bh = blockIdx.y, b_ = bh >> 4, h = bh & 15;
  const int q0 = blockIdx.x * 128;
  const int qg = q0 + wave * 16 + lrow;   // this lane's query column

  // Q fragments (B-operand), invariant across the k-loop
  v16bf qf[4];
  const __bf16* qbase = Qb + (size_t)(b_ * TS + q0 + wave * 16 + lrow) * TD + h * TDH;
#pragma unroll
  for (int c = 0; c < 4; c++) qf[c] = frag_ld(qbase + c * 32, lhi);

  v8f o[8];
#pragma unroll
  for (int dc = 0; dc < 8; dc++)
#pragma unroll
    for (int e = 0; e < 8; e++) o[dc][e] = 0.f;
  float mrun = -__builtin_inff(), lrun = 0.f;

  const int kr = tid >> 3, kc = (tid & 7) * 16;   // K-tile staging coords
  const int vr = tid >> 1, vc = (tid & 1) * 16;   // V-tile staging coords
  const int kend = q0 + 128;
  for (int k0 = 0; k0 < kend; k0 += 32) {
    __syncthreads();          // previous iteration's LDS reads complete
    {  // stage K tile: 32 keys x 128 d
      const __bf16* src = Kb + (size_t)(b_ * TS + k0 + kr) * TD + h * TDH + kc;
      cp16_g2l(src,     Ks + kr * SK + kc);
      cp16_g2l(src + 8, Ks + kr * SK + kc + 8);
    }
    {  // stage V_t tile: 128 d x 32 keys (k-contiguous rows)
      const __bf16* src = Vt + ((size_t)bh * TDH + vr) * TS + k0 + vc;
      cp16_g2l(src,     Vs + vr * SV + vc);
      cp16_g2l(src + 8, Vs + vr * SV + vc + 8);
    }
    if (k0 + 32 < kend) {     // prefetch next tiles into cache
      __builtin_prefetch(Kb + (size_t)(b_ * TS + k0 + 32 + kr) * TD + h * TDH + kc, 0, 1);
      __builtin_prefetch(Vt + ((size_t)bh * TDH + vr) * TS + k0 + 32 + vc, 0, 1);
    }
    async_wait0();
    __syncthreads();

    // scores S^T: two 16-key subtiles, each 4 WMMAs over DH=128
    float pv[2][8];
    // masking only needed when this 32-key tile can reach past a query of
    // this wave; wave-uniform condition -> scalar branch, EXEC stays full
    const bool needmask = (k0 + 31) > (q0 + wave * 16);
#pragma unroll
    for (int sub = 0; sub < 2; sub++) {
      v8f st;
#pragma unroll
      for (int e = 0; e < 8; e++) st[e] = 0.f;
#pragma unroll
      for (int c = 0; c < 4; c++) {
        v16bf a = frag_ld(Ks + (sub * 16 + lrow) * SK + c * 32, lhi);
        st = wmma_bf16(a, qf[c], st);
      }
      if (needmask) {
#pragma unroll
        for (int v = 0; v < 8; v++) {
          const int kg = k0 + sub * 16 + v + 8 * lhi;
          pv[sub][v] = (kg <= qg) ? st[v] : -__builtin_inff();
        }
      } else {
#pragma unroll
        for (int v = 0; v < 8; v++) pv[sub][v] = st[v];
      }
    }

    // online softmax per query (in-lane + one xor-16 exchange)
    float mt = -__builtin_inff();
#pragma unroll
    for (int sub = 0; sub < 2; sub++)
#pragma unroll
      for (int v = 0; v < 8; v++) mt = fmaxf(mt, pv[sub][v]);
    mt = fmaxf(mt, __shfl_xor(mt, 16, 32));
    const float mnew = fmaxf(mrun, mt);
    const float corr = __expf(mrun - mnew);
    float psum = 0.f;
#pragma unroll
    for (int sub = 0; sub < 2; sub++)
#pragma unroll
      for (int v = 0; v < 8; v++) {
        const float p = __expf(pv[sub][v] - mnew);
        pv[sub][v] = p;
        psum += p;
      }
    psum += __shfl_xor(psum, 16, 32);
    lrun = lrun * corr + psum;
    mrun = mnew;
#pragma unroll
    for (int dc = 0; dc < 8; dc++)
#pragma unroll
      for (int e = 0; e < 8; e++) o[dc][e] *= corr;

    // stage P (bf16) per wave: lane writes 8 contiguous keys -> one b128.
    // Must stay a plain DS store: we rely on same-wave DS ordering for the
    // immediate reload below (ASYNC ops would not preserve it).
#pragma unroll
    for (int sub = 0; sub < 2; sub++) {
      union { __bf16 h8[8]; uint4 q; } u;
#pragma unroll
      for (int v = 0; v < 8; v++) u.h8[v] = f2bf(pv[sub][v]);
      *(uint4*)(Ps + (wave * 16 + lrow) * SP + sub * 16 + 8 * lhi) = u.q;
    }
    v16bf pf = frag_ld(Ps + (wave * 16 + lrow) * SP, lhi);
#pragma unroll
    for (int dc = 0; dc < 8; dc++) {
      v16bf a = frag_ld(Vs + (dc * 16 + lrow) * SV, lhi);
      o[dc] = wmma_bf16(a, pf, o[dc]);
    }
  }

  const float inv = 1.f / lrun;
#pragma unroll
  for (int dc = 0; dc < 8; dc++)
#pragma unroll
    for (int v = 0; v < 8; v++) {
      const int d = dc * 16 + v + 8 * lhi;
      Ob[(size_t)(b_ * TS + qg) * TD + h * TDH + d] = f2bf(o[dc][v] * inv);
    }
}

// ---------- host-side orchestration ----------
extern "C" void kernel_launch(void* const* d_in, const int* in_sizes, int n_in,
                              void* d_out, int out_size, void* d_ws, size_t ws_size,
                              hipStream_t stream) {
  (void)in_sizes; (void)n_in; (void)out_size; (void)ws_size;
  const float* X    = (const float*)d_in[0];
  const float* Wq   = (const float*)d_in[1];
  const float* bq   = (const float*)d_in[2];
  const float* Wk   = (const float*)d_in[3];
  const float* bk   = (const float*)d_in[4];
  const float* Wv   = (const float*)d_in[5];
  const float* bv   = (const float*)d_in[6];
  const float* Wo   = (const float*)d_in[7];
  const float* bo   = (const float*)d_in[8];
  const float* Win  = (const float*)d_in[9];
  const float* bin  = (const float*)d_in[10];
  const float* Wout = (const float*)d_in[11];
  const float* bout = (const float*)d_in[12];
  const float* ln1w = (const float*)d_in[13];
  const float* ln1b = (const float*)d_in[14];
  const float* ln2w = (const float*)d_in[15];
  const float* ln2b = (const float*)d_in[16];
  float* out = (float*)d_out;

  char* ws = (char*)d_ws;
  const size_t szMD = (size_t)TM * TD * 2;          // 16.78 MB (bf16 M x D)
  __bf16* lnb = (__bf16*)(ws);                      // LN output (reused for LN2)
  __bf16* Qb  = (__bf16*)(ws + 1 * szMD);
  __bf16* Kb  = (__bf16*)(ws + 2 * szMD);
  __bf16* Vt  = (__bf16*)(ws + 3 * szMD);
  __bf16* Ab  = (__bf16*)(ws + 4 * szMD);           // attention output
  float*  X1  = (float*)(ws + 5 * szMD);            // residual stream (f32, 2*szMD)
  __bf16* Hm  = (__bf16*)(ws + 1 * szMD);           // MLP hidden reuses Qb..Ab (4*szMD)

  const float qk = 0.29730177875068026f;            // 1/sqrt(sqrt(128))
  dim3 th(256);
  dim3 gD(TD / 128, TM / 128);      // (16, 32)
  dim3 gF(TDFF / 128, TM / 128);    // (64, 32)
  dim3 gA(TS / 128, TB * THD);      // (16, 32)

  layernorm_bf16_k<<<TM, th, 0, stream>>>(X, ln1w, ln1b, lnb);
  gemm_bf16_k<0, false, false><<<gD, th, 0, stream>>>(lnb, Wq, bq, nullptr, Qb, TM, TD, TD, qk);
  gemm_bf16_k<0, false, false><<<gD, th, 0, stream>>>(lnb, Wk, bk, nullptr, Kb, TM, TD, TD, qk);
  gemm_bf16_k<2, false, false><<<gD, th, 0, stream>>>(lnb, Wv, bv, nullptr, Vt, TM, TD, TD, 1.f);
  attn_k<<<gA, th, 0, stream>>>(Qb, Kb, Vt, Ab);
  gemm_bf16_k<1, false, true><<<gD, th, 0, stream>>>(Ab, Wo, bo, X, (void*)X1, TM, TD, TD, 1.f);
  layernorm_bf16_k<<<TM, th, 0, stream>>>(X1, ln2w, ln2b, lnb);
  gemm_bf16_k<0, true, false><<<gF, th, 0, stream>>>(lnb, Win, bin, nullptr, Hm, TM, TDFF, TD, 1.f);
  gemm_bf16_k<1, false, true><<<gD, th, 0, stream>>>(Hm, Wout, bout, X1, (void*)out, TM, TD, TDFF, 1.f);
}
